// AsynchronousGRUActor_60722247631600
// MI455X (gfx1250) — matrix operand order repaired and checked by reading, not compiled, not run on hardware
//
#include <hip/hip_runtime.h>
#include <hip/hip_bf16.h>

// ---- problem constants (match reference) ----
static constexpr int OBS = 128;
static constexpr int HID = 64;
static constexpr int NA  = 16;     // action dim
static constexpr int WAVES = 8;
static constexpr int TPB = WAVES * 32;

typedef __attribute__((ext_vector_type(16))) __bf16        v16bf;
typedef __attribute__((ext_vector_type(8)))  float         v8f;
typedef __attribute__((ext_vector_type(8)))  unsigned int  v8u;

// ---- exact RNE f32->bf16 (used once, in the weight repack kernel) ----
__device__ __forceinline__ unsigned short bf16_rne(float a) {
    unsigned int ua = __float_as_uint(a);
    ua += 0x7FFFu + ((ua >> 16) & 1u);
    return (unsigned short)(ua >> 16);
}

// ---- fast f32 pair -> packed bf16 (truncation): single v_perm_b32 ----
// dst = { hi16(b), hi16(a) }  (a = low half). Selector picks bytes
// {b.3, b.2, a.3, a.2} with src0=b (bytes 4..7), src1=a (bytes 0..3).
#if __has_builtin(__builtin_amdgcn_perm)
__device__ __forceinline__ unsigned int pk_bf16(float a, float b) {
    return __builtin_amdgcn_perm(__float_as_uint(b), __float_as_uint(a),
                                 0x07060302u);
}
#else
__device__ __forceinline__ unsigned int pk_bf16(float a, float b) {
    return (__float_as_uint(a) >> 16) | (__float_as_uint(b) & 0xFFFF0000u);
}
#endif

// ---- hardware tanh (gfx1250 v_tanh_f32) with libm fallback ----
#if __has_builtin(__builtin_amdgcn_tanhf)
__device__ __forceinline__ float fast_tanh(float x) { return __builtin_amdgcn_tanhf(x); }
#else
__device__ __forceinline__ float fast_tanh(float x) { return tanhf(x); }
#endif

// sigmoid(x) = 0.5*tanh(x/2) + 0.5  -> single trans op
__device__ __forceinline__ float fast_sigmoid(float x) {
    return __builtin_fmaf(0.5f, fast_tanh(0.5f * x), 0.5f);
}

// A-matrix fragment (16x32 bf16, MxK) for v_wmma_f32_16x16x32_bf16.
// Per lane: row M = lane&15; elements 0..15 = K[k0 .. k0+15],
//           where k0 = kBase + (lane>=16)*16.
// Caller passes rowp = &src[M * stride] and k0.
__device__ __forceinline__ v16bf load_afrag_f32(const float* rowp, int k0) {
    float4 f0 = *(const float4*)(rowp + k0);
    float4 f1 = *(const float4*)(rowp + k0 + 4);
    float4 f2 = *(const float4*)(rowp + k0 + 8);
    float4 f3 = *(const float4*)(rowp + k0 + 12);
    v8u u;
    u[0] = pk_bf16(f0.x, f0.y); u[1] = pk_bf16(f0.z, f0.w);
    u[2] = pk_bf16(f1.x, f1.y); u[3] = pk_bf16(f1.z, f1.w);
    u[4] = pk_bf16(f2.x, f2.y); u[5] = pk_bf16(f2.z, f2.w);
    u[6] = pk_bf16(f3.x, f3.y); u[7] = pk_bf16(f3.z, f3.w);
    return __builtin_bit_cast(v16bf, u);
}

// B-matrix fragment (32x16 bf16, KxN). Per lane: column N = n (caller passes
// n = nBase + (lane&15)); elements 0..15 = K[k0 .. k0+15] with
// k0 = kBase + (lane>=16)*16. Weights stored bf16 as [n][k] row-major.
__device__ __forceinline__ v16bf load_bfrag(const unsigned short* w, int n,
                                            int stride, int k0) {
    const uint4* p = (const uint4*)(w + n * stride + k0);
    uint4 lo = p[0];
    uint4 hi = p[1];
    v8u u;
    u[0] = lo.x; u[1] = lo.y; u[2] = lo.z; u[3] = lo.w;
    u[4] = hi.x; u[5] = hi.y; u[6] = hi.z; u[7] = hi.w;
    return __builtin_bit_cast(v16bf, u);
}

__device__ __forceinline__ v8f wmma_bf16(v16bf a, v16bf b, v8f c) {
    return __builtin_amdgcn_wmma_f32_16x16x32_bf16(
        /*neg_a=*/false, a, /*neg_b=*/false, b,
        /*c_mod=*/(short)0, c, /*reuse_a=*/false, /*reuse_b=*/false);
}

__device__ __forceinline__ v8f splat8(float s) {
    v8f v;
    #pragma unroll
    for (int e = 0; e < 8; ++e) v[e] = s;
    return v;
}

// ---- kernel 0: fp32 -> bf16 weight repack into workspace (stays L2-resident) ----
__global__ void cvt_weights_kernel(const float* __restrict__ fc1_w,
                                   const float* __restrict__ w_ih,
                                   const float* __restrict__ w_hh,
                                   const float* __restrict__ fc2_w,
                                   unsigned short* __restrict__ ws) {
    const int N1 = HID * OBS;            // fc1_w  [64][128]
    const int N2 = N1 + 3 * HID * HID;   // w_ih   [192][64]
    const int N3 = N2 + 3 * HID * HID;   // w_hh   [192][64]
    const int N4 = N3 + NA * HID;        // fc2_w  [16][64]
    int i = blockIdx.x * blockDim.x + threadIdx.x;
    if (i >= N4) return;
    float v;
    if      (i < N1) v = fc1_w[i];
    else if (i < N2) v = w_ih[i - N1];
    else if (i < N3) v = w_hh[i - N2];
    else             v = fc2_w[i - N3];
    ws[i] = bf16_rne(v);
}

// ---- kernel 1: fused fc1+tanh -> GRU cell -> masked update -> fc2 ----
__global__ void __launch_bounds__(TPB)
gru_actor_kernel(const float* __restrict__ x, const float* __restrict__ hid,
                 const int* __restrict__ mask,
                 const float* __restrict__ fc1_b, const float* __restrict__ b_ih,
                 const float* __restrict__ b_hh, const float* __restrict__ fc2_b,
                 const unsigned short* __restrict__ ws,
                 float* __restrict__ out, int Btot) {
    __shared__ float stageS[WAVES][16 * HID];   // a, then new_hidden (per wave)
    __shared__ float lstageS[WAVES][16 * NA];   // logits tile (per wave)

    const int lane = threadIdx.x & 31;
    const int wave = threadIdx.x >> 5;
    const int ln   = lane & 15;    // N index / A-row index within tile
    const int hs   = lane >> 4;    // half-wave select

    const int tile    = blockIdx.x * WAVES + wave;   // uniform per wave
    const int rowBase = tile * 16;
    if (rowBase >= Btot) return;                     // wave-uniform: EXEC stays all-1s

    float* stage  = stageS[wave];
    float* lstage = lstageS[wave];

    const unsigned short* ws_fc1 = ws;
    const unsigned short* ws_wih = ws + HID * OBS;
    const unsigned short* ws_whh = ws + HID * OBS + 3 * HID * HID;
    const unsigned short* ws_fc2 = ws + HID * OBS + 6 * HID * HID;

    const int rowA = min(rowBase + ln, Btot - 1);    // A-fragment source row

    // ================= Phase 1: a = tanh(x @ fc1_w.T + fc1_b) =================
    v16bf xA[4];
    {
        const float* xrow = x + (size_t)rowA * OBS;
        #pragma unroll
        for (int kt = 0; kt < 4; ++kt)
            xA[kt] = load_afrag_f32(xrow, kt * 32 + hs * 16);
    }
    #pragma unroll
    for (int j = 0; j < 4; ++j) {                    // 4 x 16 output columns
        v8f acc = splat8(fc1_b[j * 16 + ln]);
        #pragma unroll
        for (int kt = 0; kt < 4; ++kt)
            acc = wmma_bf16(xA[kt],
                            load_bfrag(ws_fc1, j * 16 + ln, OBS, kt * 32 + hs * 16),
                            acc);
        #pragma unroll
        for (int e = 0; e < 8; ++e)                  // C/D layout: M = e + 8*hs, N = j*16+ln
            stage[(e + hs * 8) * HID + j * 16 + ln] = fast_tanh(acc[e]);
    }

    // ================= Phase 2: GRU cell =================
    v16bf aA[2], hA[2];
    const float* hrow = hid + (size_t)rowA * HID;
    #pragma unroll
    for (int kt = 0; kt < 2; ++kt) {
        aA[kt] = load_afrag_f32(stage + ln * HID, kt * 32 + hs * 16);
        hA[kt] = load_afrag_f32(hrow, kt * 32 + hs * 16);
    }
    int mk[8];
    #pragma unroll
    for (int e = 0; e < 8; ++e)
        mk[e] = mask[min(rowBase + e + hs * 8, Btot - 1)];

    #pragma unroll
    for (int j = 0; j < 4; ++j) {
        const int cr = j * 16 + ln;        // r-gate column
        const int cz = 64 + cr;            // z-gate column
        const int cn = 128 + cr;           // n-gate column
        // r/z: a@Wih and h@Whh share one accumulator (bias pre-summed);
        // n-gate needs i_n and h_n separately for tanh(i_n + r*h_n).
        v8f racc  = splat8(b_ih[cr] + b_hh[cr]);
        v8f zacc  = splat8(b_ih[cz] + b_hh[cz]);
        v8f inacc = splat8(b_ih[cn]);
        v8f hnacc = splat8(b_hh[cn]);
        #pragma unroll
        for (int kt = 0; kt < 2; ++kt) {
            const int k0 = kt * 32 + hs * 16;
            racc  = wmma_bf16(aA[kt], load_bfrag(ws_wih, cr, HID, k0), racc);
            racc  = wmma_bf16(hA[kt], load_bfrag(ws_whh, cr, HID, k0), racc);
            zacc  = wmma_bf16(aA[kt], load_bfrag(ws_wih, cz, HID, k0), zacc);
            zacc  = wmma_bf16(hA[kt], load_bfrag(ws_whh, cz, HID, k0), zacc);
            inacc = wmma_bf16(aA[kt], load_bfrag(ws_wih, cn, HID, k0), inacc);
            hnacc = wmma_bf16(hA[kt], load_bfrag(ws_whh, cn, HID, k0), hnacc);
        }
        #pragma unroll
        for (int e = 0; e < 8; ++e) {
            const int M    = e + hs * 8;
            const int grow = min(rowBase + M, Btot - 1);
            float ho = hid[(size_t)grow * HID + j * 16 + ln];
            float r  = fast_sigmoid(racc[e]);
            float z  = fast_sigmoid(zacc[e]);
            float nn = fast_tanh(inacc[e] + r * hnacc[e]);
            float hu = (1.0f - z) * nn + z * ho;
            stage[M * HID + j * 16 + ln] = mk[e] ? hu : ho;   // new_hidden tile
        }
    }

    // ================= Phase 3: logits = new_hidden @ fc2_w.T + fc2_b =================
    {
        v8f lacc = splat8(fc2_b[ln]);
        #pragma unroll
        for (int kt = 0; kt < 2; ++kt) {
            v16bf hnA = load_afrag_f32(stage + ln * HID, kt * 32 + hs * 16);
            lacc = wmma_bf16(hnA, load_bfrag(ws_fc2, ln, HID, kt * 32 + hs * 16), lacc);
        }
        #pragma unroll
        for (int e = 0; e < 8; ++e)
            lstage[(e + hs * 8) * NA + ln] = lacc[e];
    }

    // ================= Coalesced output stores =================
    float* outLog = out;
    float* outHid = out + (size_t)Btot * NA;
    const float4* lst4 = (const float4*)lstage;
    #pragma unroll
    for (int q = 0; q < 2; ++q) {            // 16x16 logits tile = 64 float4
        int f4  = q * 32 + lane;
        int row = f4 >> 2, c4 = f4 & 3;
        if (rowBase + row < Btot)
            *(float4*)(outLog + (size_t)(rowBase + row) * NA + c4 * 4) = lst4[f4];
    }
    const float4* st4 = (const float4*)stage;
    #pragma unroll
    for (int q = 0; q < 8; ++q) {            // 16x64 hidden tile = 256 float4
        int f4  = q * 32 + lane;
        int row = f4 >> 4, c4 = f4 & 15;
        if (rowBase + row < Btot)
            *(float4*)(outHid + (size_t)(rowBase + row) * HID + c4 * 4) = st4[f4];
    }
}

extern "C" void kernel_launch(void* const* d_in, const int* in_sizes, int n_in,
                              void* d_out, int out_size, void* d_ws, size_t ws_size,
                              hipStream_t stream) {
    const float* x     = (const float*)d_in[0];
    const float* hid   = (const float*)d_in[1];
    const int*   mask  = (const int*)d_in[2];
    const float* fc1_w = (const float*)d_in[3];
    const float* fc1_b = (const float*)d_in[4];
    const float* w_ih  = (const float*)d_in[5];
    const float* w_hh  = (const float*)d_in[6];
    const float* b_ih  = (const float*)d_in[7];
    const float* b_hh  = (const float*)d_in[8];
    const float* fc2_w = (const float*)d_in[9];
    const float* fc2_b = (const float*)d_in[10];
    float* out = (float*)d_out;
    unsigned short* ws = (unsigned short*)d_ws;

    const int Btot = in_sizes[0] / OBS;

    const int ncvt = HID * OBS + 6 * HID * HID + NA * HID;   // 33792 bf16 weights
    cvt_weights_kernel<<<(ncvt + 255) / 256, 256, 0, stream>>>(fc1_w, w_ih, w_hh,
                                                               fc2_w, ws);

    const int nTiles  = (Btot + 15) / 16;
    const int nBlocks = (nTiles + WAVES - 1) / WAVES;
    gru_actor_kernel<<<nBlocks, TPB, 0, stream>>>(x, hid, mask, fc1_b, b_ih, b_hh,
                                                  fc2_b, ws, out, Btot);
}
